// _LSTM_22454089023921
// MI455X (gfx1250) — matrix operand order repaired and checked by reading, compiled, not run on hardware
//
#include <hip/hip_runtime.h>

// ---------------- problem constants ----------------
#define BB    32768
#define TTT   276
#define HH    8
#define G4H   32              // 4*H
#define KDIM  (TTT * 2 * HH)  // 4416 = flattened h width
#define NOUT  TTT             // 276 FC outputs
#define NPAD  288             // padded N (18 full 16-tiles)
#define KSTEPS (KDIM / 32)    // 138 exact (even -> clean ping-pong)
#define NT_PER_WAVE 3
#define NGROUPS 6             // 6*3*16 = 288

typedef __attribute__((ext_vector_type(16))) __bf16 v16bf;
typedef __attribute__((ext_vector_type(8)))  float  v8f;

// ---------------- helpers ----------------
__device__ __forceinline__ float fast_exp(float x) {
    return __builtin_amdgcn_exp2f(x * 1.44269504088896f);   // v_exp_f32
}
// gate magnitudes are bounded (|bias|+|Wih||x|+|Whh||h| < ~8 for x~N(0,1)); no clamp needed
__device__ __forceinline__ float sigm(float x) {
    return __builtin_amdgcn_rcpf(1.0f + fast_exp(-x));      // v_rcp_f32
}
// c can grow over 276 steps -> clamp to avoid exp2 overflow -> NaN
__device__ __forceinline__ float fast_tanh(float x) {
    x = fminf(fmaxf(x, -15.0f), 15.0f);
    float e = fast_exp(-2.0f * x);
    return (1.0f - e) * __builtin_amdgcn_rcpf(1.0f + e);
}
__device__ __forceinline__ unsigned short f2bfu(float f) {   // RNE f32->bf16 bits
    unsigned u = __builtin_bit_cast(unsigned, f);
    return (unsigned short)((u + 0x7FFFu + ((u >> 16) & 1u)) >> 16);
}

// ---------------- kernel 1: Wfc f32 -> bf16, zero-padded to NPAD rows ----------------
__global__ void cvt_wfc_bf16(const float* __restrict__ src, __bf16* __restrict__ dst,
                             int ntot, int nsrc) {
    int i = blockIdx.x * blockDim.x + threadIdx.x;
    if (i < ntot) {
        unsigned short s = (i < nsrc) ? f2bfu(src[i]) : (unsigned short)0;
        dst[i] = __builtin_bit_cast(__bf16, s);
    }
}

// ---------------- kernel 2: x [B,T,2] -> xT [T,B,2] (LDS-tiled) ----------------
__global__ void transpose_x(const float2* __restrict__ x, float2* __restrict__ xT) {
    __shared__ float2 tile[32][33];
    const int bt = blockIdx.x * 32;   // t tile base
    const int bb = blockIdx.y * 32;   // b tile base
    const int lx = threadIdx.x;       // 0..31
    const int ly = threadIdx.y;       // 0..7
#pragma unroll
    for (int i = 0; i < 32; i += 8) {
        int t = bt + lx;
        int b = bb + ly + i;
        if (t < TTT) tile[ly + i][lx] = x[(size_t)b * TTT + t];
    }
    __syncthreads();
#pragma unroll
    for (int i = 0; i < 32; i += 8) {
        int t = bt + ly + i;
        int b = bb + lx;
        if (t < TTT) xT[(size_t)t * BB + b] = tile[lx][ly + i];
    }
}

// ---------------- kernel 3: fused bidirectional LSTM ----------------
// thread = (batch, dir); h/c in registers; weights in LDS (broadcast b128 reads);
// x read coalesced from xT; h written as packed bf16 (16B store) in GEMM-A layout.
__global__ void lstm_kernel(const float2* __restrict__ xT,
                            const float* __restrict__ Wih_f, const float* __restrict__ Whh_f,
                            const float* __restrict__ bih_f, const float* __restrict__ bhh_f,
                            const float* __restrict__ Wih_b, const float* __restrict__ Whh_b,
                            const float* __restrict__ bih_b, const float* __restrict__ bhh_b,
                            __bf16* __restrict__ flatA)
{
    __shared__ __align__(16) float  sWhh[G4H][HH];
    __shared__ __align__(16) float4 sGate[G4H];   // {bias, wih0, wih1, 0} per gate

    const int dir = blockIdx.y;
    const float* Wih = dir ? Wih_b : Wih_f;
    const float* Whh = dir ? Whh_b : Whh_f;
    const float* bih = dir ? bih_b : bih_f;
    const float* bhh = dir ? bhh_b : bhh_f;

    const int tid = threadIdx.x;
    if (tid < G4H) {
        float4 g;
        g.x = bih[tid] + bhh[tid];
        g.y = Wih[tid * 2 + 0];
        g.z = Wih[tid * 2 + 1];
        g.w = 0.0f;
        sGate[tid] = g;
#pragma unroll
        for (int j = 0; j < HH; ++j) sWhh[tid][j] = Whh[tid * HH + j];
    }
    __syncthreads();

    const int b = blockIdx.x * blockDim.x + tid;
    __bf16* ob = flatA + (size_t)b * KDIM + dir * HH;

    float h[HH], c[HH];
#pragma unroll
    for (int j = 0; j < HH; ++j) { h[j] = 0.0f; c[j] = 0.0f; }

    for (int s = 0; s < TTT; ++s) {
        const int t = dir ? (TTT - 1 - s) : s;
        const float2 xv = xT[(size_t)t * BB + b];   // coalesced 256B/wave
        const float x0 = xv.x, x1 = xv.y;

        float hn[HH];
#pragma unroll
        for (int j = 0; j < HH; ++j) {
            const float4 pi = sGate[j];
            const float4 pf = sGate[HH + j];
            const float4 pg = sGate[2 * HH + j];
            const float4 po = sGate[3 * HH + j];
            float gi = pi.x + pi.y * x0 + pi.z * x1;
            float gf = pf.x + pf.y * x0 + pf.z * x1;
            float gg = pg.x + pg.y * x0 + pg.z * x1;
            float go = po.x + po.y * x0 + po.z * x1;
#pragma unroll
            for (int k = 0; k < HH; ++k) {
                gi += sWhh[j][k]          * h[k];
                gf += sWhh[HH + j][k]     * h[k];
                gg += sWhh[2 * HH + j][k] * h[k];
                go += sWhh[3 * HH + j][k] * h[k];
            }
            const float iv = sigm(gi);
            const float fv = sigm(gf);
            const float gv = fast_tanh(gg);
            const float ov = sigm(go);
            c[j]  = fv * c[j] + iv * gv;
            hn[j] = ov * fast_tanh(c[j]);
        }

        unsigned short pk[8];
#pragma unroll
        for (int j = 0; j < HH; ++j) { h[j] = hn[j]; pk[j] = f2bfu(hn[j]); }
        // 16-byte aligned packed store of 8 bf16
        *reinterpret_cast<uint4*>(ob + (size_t)t * (2 * HH)) = *reinterpret_cast<const uint4*>(pk);
    }
}

// ---------------- kernel 4: FC GEMM via v_wmma_f32_16x16x32_bf16 ----------------
// out[b, n] = sum_k flatA[b, k] * WfcB[n, k] + bfc[n]
// One wave computes a 16(M) x 48(N = 3 tiles) strip.
// Double-buffered K loop with PEELED last iteration: steady-state body is
// branch-free (load odd -> wmma even -> load even -> wmma odd), so registers
// rotate cleanly (no acc copies / phi merges) and waits stay partial.
__global__ void __launch_bounds__(256)
fc_wmma_kernel(const __bf16* __restrict__ A,   // [BB, KDIM] bf16
               const __bf16* __restrict__ Bw,  // [NPAD, KDIM] bf16
               const float*  __restrict__ bfc, // [NOUT]
               float*        __restrict__ out) // [BB, NOUT]
{
    const int wave  = (blockIdx.x * blockDim.x + threadIdx.x) >> 5;
    const int lane  = threadIdx.x & 31;
    const int mtile = wave / NGROUPS;          // 0 .. 2047
    const int ngrp  = wave % NGROUPS;          // 0 .. 5

    const int m0   = mtile * 16;
    const int half = lane >> 4;                // 0: lanes 0-15, 1: lanes 16-31
    const int r    = lane & 15;

    const __bf16* arow  = A + (size_t)(m0 + r) * KDIM;
    const __bf16* brow0 = Bw + (size_t)(ngrp * (NT_PER_WAVE * 16) + r) * KDIM;

    union V16 { v16bf v; uint4 q[2]; };

    // A 16x32 bf16 tile: lanes 0-15 K=[0..8)+[16..24), lanes 16-31 K=[8..16)+[24..32)
    // B 32x16 bf16 tile: lane&15 = column, same K split as A
#define LOAD_AB(aa, bb, k0)                                                        \
    do {                                                                           \
        (aa).q[0] = *reinterpret_cast<const uint4*>(arow + (k0) + half * 8);       \
        (aa).q[1] = *reinterpret_cast<const uint4*>(arow + (k0) + 16 + half * 8);  \
        _Pragma("unroll")                                                          \
        for (int nt = 0; nt < NT_PER_WAVE; ++nt) {                                 \
            const __bf16* brow = brow0 + (size_t)nt * 16 * KDIM + (k0);            \
            (bb)[nt].q[0] = *reinterpret_cast<const uint4*>(brow + half * 8);      \
            (bb)[nt].q[1] = *reinterpret_cast<const uint4*>(brow + 16 + half * 8); \
        }                                                                          \
    } while (0)

#define WMMA_STEP(aa, bb)                                                          \
    do {                                                                           \
        _Pragma("unroll")                                                          \
        for (int nt = 0; nt < NT_PER_WAVE; ++nt)                                   \
            acc[nt] = __builtin_amdgcn_wmma_f32_16x16x32_bf16(                     \
                false, (aa).v, false, (bb)[nt].v, (short)0, acc[nt], false, false);\
    } while (0)

    v8f zero8 = {};
    v8f acc[NT_PER_WAVE];
#pragma unroll
    for (int i = 0; i < NT_PER_WAVE; ++i) acc[i] = zero8;

    V16 a0, a1;
    V16 b0[NT_PER_WAVE], b1[NT_PER_WAVE];

    LOAD_AB(a0, b0, 0);
    // steady state: branch-free body
    for (int ks = 0; ks < KSTEPS - 2; ks += 2) {
        LOAD_AB(a1, b1, (ks + 1) * 32);          // prefetch odd step
        WMMA_STEP(a0, b0);                        // consume even step
        LOAD_AB(a0, b0, (ks + 2) * 32);          // prefetch next even step
        WMMA_STEP(a1, b1);                        // consume odd step
    }
    // peeled final pair (ks = KSTEPS-2, KSTEPS-1)
    LOAD_AB(a1, b1, (KSTEPS - 1) * 32);
    WMMA_STEP(a0, b0);
    WMMA_STEP(a1, b1);
#undef LOAD_AB
#undef WMMA_STEP

    // C/D layout: lane&15 = N column, VGPR rr = M row (lanes 16-31 -> M += 8)
#pragma unroll
    for (int nt = 0; nt < NT_PER_WAVE; ++nt) {
        const int n = ngrp * (NT_PER_WAVE * 16) + nt * 16 + r;
        if (n < NOUT) {
            const float bias = bfc[n];
#pragma unroll
            for (int rr = 0; rr < 8; ++rr) {
                const int row = m0 + half * 8 + rr;
                out[(size_t)row * NOUT + n] = acc[nt][rr] + bias;
            }
        }
    }
}

// ---------------- launch ----------------
extern "C" void kernel_launch(void* const* d_in, const int* in_sizes, int n_in,
                              void* d_out, int out_size, void* d_ws, size_t ws_size,
                              hipStream_t stream) {
    const float* x     = (const float*)d_in[0];
    const float* Wih_f = (const float*)d_in[1];
    const float* Whh_f = (const float*)d_in[2];
    const float* bih_f = (const float*)d_in[3];
    const float* bhh_f = (const float*)d_in[4];
    const float* Wih_b = (const float*)d_in[5];
    const float* Whh_b = (const float*)d_in[6];
    const float* bih_b = (const float*)d_in[7];
    const float* bhh_b = (const float*)d_in[8];
    const float* Wfc   = (const float*)d_in[9];
    const float* bfc   = (const float*)d_in[10];
    float* out = (float*)d_out;

    // workspace layout (all 16B aligned):
    //   [flatA bf16: BB*KDIM] [WfcB bf16: NPAD*KDIM] [xT f32: TTT*BB*2]
    char* ws = (char*)d_ws;
    __bf16* flatA = (__bf16*)ws;
    __bf16* WfcB  = (__bf16*)(ws + (size_t)BB * KDIM * sizeof(__bf16));
    float2* xT    = (float2*)(ws + (size_t)BB * KDIM * sizeof(__bf16)
                                 + (size_t)NPAD * KDIM * sizeof(__bf16));

    // 1) Wfc -> bf16, zero-padded to 288 rows (no guards in GEMM K loop)
    {
        const int ntot = NPAD * KDIM;
        const int nsrc = NOUT * KDIM;
        cvt_wfc_bf16<<<(ntot + 255) / 256, 256, 0, stream>>>(Wfc, WfcB, ntot, nsrc);
    }
    // 2) x [B,T,2] -> xT [T,B,2] so LSTM x loads are lane-coalesced
    transpose_x<<<dim3((TTT + 31) / 32, BB / 32), dim3(32, 8), 0, stream>>>(
        (const float2*)x, xT);
    // 3) bidirectional LSTM, writes flatA in WMMA-A bf16 layout
    lstm_kernel<<<dim3(BB / 256, 2), 256, 0, stream>>>(
        xT, Wih_f, Whh_f, bih_f, bhh_f, Wih_b, Whh_b, bih_b, bhh_b, flatA);
    // 4) FC GEMM with bf16 WMMA (double-buffered, peeled epilogue)
    {
        const int waves  = (BB / 16) * NGROUPS;   // 12288
        const int blocks = waves / 8;             // 256 threads = 8 waves per block
        fc_wmma_kernel<<<blocks, 256, 0, stream>>>(flatA, WfcB, bfc, out);
    }
}
